// TuckERCPD_49254684951151
// MI455X (gfx1250) — matrix operand order repaired and checked by reading, not compiled, not run on hardware
//
#include <hip/hip_runtime.h>
#include <stdint.h>

// Problem constants (from reference)
#define NE_   100000
#define D_    200      // DE = DR = DT
#define P_    200
#define B_    1024
#define KPAD  224      // 200 padded to 7 * 32 (bf16 WMMA K)
#define NTILE 128      // entities per workgroup tile
#define MTILE 128      // batch rows per m-iteration
#define LPITCH 232     // LDS row pitch in bf16 elems (16B-aligned, not pow2 stride)

typedef __attribute__((ext_vector_type(16))) __bf16 v16bf;
typedef __attribute__((ext_vector_type(8)))  __bf16 v8bf;
typedef __attribute__((ext_vector_type(8)))  float  v8f;
typedef __attribute__((ext_vector_type(4)))  float  v4f;

static __device__ inline uint16_t f2bf_rn(float x) {
  uint32_t u = __builtin_bit_cast(uint32_t, x);
  u += 0x7FFFu + ((u >> 16) & 1u);               // round-to-nearest-even
  return (uint16_t)(u >> 16);
}
static __device__ inline float bf2f(uint16_t h) {
  uint32_t u = ((uint32_t)h) << 16;
  return __builtin_bit_cast(float, u);
}

// ---------------------------------------------------------------------------
// Stage 1: h = (R[r]@F0) * (E[e1]@F1) * (T[t]@F3);  g = h @ F2^T
// Store g as split-bf16 (hi + lo) padded to KPAD. One block per batch item.
// ---------------------------------------------------------------------------
__global__ __launch_bounds__(KPAD) void tuckercpd_prep(
    const int* __restrict__ e1_idx, const int* __restrict__ r_idx,
    const int* __restrict__ t_idx,
    const float* __restrict__ E, const float* __restrict__ R,
    const float* __restrict__ T,
    const float* __restrict__ F0, const float* __restrict__ F1,
    const float* __restrict__ F2, const float* __restrict__ F3,
    uint16_t* __restrict__ gHi, uint16_t* __restrict__ gLo)
{
  __shared__ float eRow[D_], rRow[D_], tRow[D_], hRow[P_];
  const int b = blockIdx.x;
  const int t = threadIdx.x;
  const int ie = e1_idx[b], ir = r_idx[b], it = t_idx[b];
  if (t < D_) {
    eRow[t] = E[(size_t)ie * D_ + t];
    rRow[t] = R[(size_t)ir * D_ + t];
    tRow[t] = T[(size_t)it * D_ + t];
  }
  __syncthreads();
  if (t < P_) {
    float fr = 0.f, fe = 0.f, ft = 0.f;
    for (int k = 0; k < D_; ++k) {
      fr = fmaf(rRow[k], F0[k * P_ + t], fr);
      fe = fmaf(eRow[k], F1[k * P_ + t], fe);
      ft = fmaf(tRow[k], F3[k * P_ + t], ft);
    }
    hRow[t] = fr * fe * ft;
  }
  __syncthreads();
  // g[b,k] = sum_p h[p] * F2[k,p]   (k = threadIdx; k >= D_ -> zero pad)
  float gv = 0.f;
  if (t < D_) {
    for (int p = 0; p < P_; ++p) gv = fmaf(hRow[p], F2[t * P_ + p], gv);
  }
  const uint16_t hi = f2bf_rn(gv);
  const uint16_t lo = f2bf_rn(gv - bf2f(hi));
  gHi[(size_t)b * KPAD + t] = hi;
  gLo[(size_t)b * KPAD + t] = lo;
}

// ---------------------------------------------------------------------------
// Stage 2: out = sigmoid(g @ E^T), (1024 x 100000) fp32.
// One WG per 128-entity tile; E tile split to bf16 hi/lo in LDS once, then
// all 8 batch tiles are computed (E read from HBM exactly once).
// Split-bf16 product: hi*hi + hi*lo + lo*hi  (3x v_wmma_f32_16x16x32_bf16).
// Output streamed with nontemporal stores (write-once, keep L2 for g/A).
// ---------------------------------------------------------------------------
__global__ __launch_bounds__(256) void tuckercpd_gemm(
    const float* __restrict__ E,
    const uint16_t* __restrict__ gHi, const uint16_t* __restrict__ gLo,
    float* __restrict__ out)
{
  __shared__ uint16_t ldsHi[NTILE * LPITCH];
  __shared__ uint16_t ldsLo[NTILE * LPITCH];

  const int tid   = threadIdx.x;
  const int tileN = blockIdx.x * NTILE;

  // ---- Fill LDS: 128 rows x 50 float4 of E, converted to bf16 hi/lo ----
  for (int i = 0; i < 25; ++i) {
    const int idx = i * 256 + tid;             // 6400 float4s total
    const int row = idx / 50;
    const int c4  = idx % 50;
    const int n   = tileN + row;
    v4f v = {0.f, 0.f, 0.f, 0.f};
    if (n < NE_) {
      // E lines are touched by exactly one workgroup: non-temporal load.
      v = __builtin_nontemporal_load(
          reinterpret_cast<const v4f*>(E + (size_t)n * D_ + c4 * 4));
    }
    uint32_t hw[2], lw[2];
#pragma unroll
    for (int j = 0; j < 2; ++j) {
      const uint16_t h0 = f2bf_rn(v[2 * j]);
      const uint16_t l0 = f2bf_rn(v[2 * j] - bf2f(h0));
      const uint16_t h1 = f2bf_rn(v[2 * j + 1]);
      const uint16_t l1 = f2bf_rn(v[2 * j + 1] - bf2f(h1));
      hw[j] = (uint32_t)h0 | ((uint32_t)h1 << 16);
      lw[j] = (uint32_t)l0 | ((uint32_t)l1 << 16);
    }
    const int e0 = row * LPITCH + c4 * 4;      // bf16 index, 8B aligned
    *reinterpret_cast<uint2*>(&ldsHi[e0]) = make_uint2(hw[0], hw[1]);
    *reinterpret_cast<uint2*>(&ldsLo[e0]) = make_uint2(lw[0], lw[1]);
  }
  // zero-pad k in [200, 224): 128 rows * 6 uint2
  for (int i = 0; i < 3; ++i) {
    const int idx = i * 256 + tid;
    const int row = idx / 6, q = idx % 6;
    const int e0 = row * LPITCH + 200 + q * 4;
    *reinterpret_cast<uint2*>(&ldsHi[e0]) = make_uint2(0u, 0u);
    *reinterpret_cast<uint2*>(&ldsLo[e0]) = make_uint2(0u, 0u);
  }
  __syncthreads();

  const int wave = tid >> 5;
  const int lane = tid & 31;
  const int half = lane >> 4;   // which 16-lane half
  const int l16  = lane & 15;
  const int wm   = wave & 1;    // 2 waves over M: 64 rows each
  const int wn   = wave >> 1;   // 4 waves over N: 32 cols each

  const v8f vzero = {0.f, 0.f, 0.f, 0.f, 0.f, 0.f, 0.f, 0.f};

  for (int mt = 0; mt < B_ / MTILE; ++mt) {
    v8f acc[4][2];
#pragma unroll
    for (int s = 0; s < 4; ++s)
#pragma unroll
      for (int u = 0; u < 2; ++u) acc[s][u] = vzero;

    for (int kc = 0; kc < KPAD / 32; ++kc) {
      // A fragments (g rows): ISA 16-bit A 16x32 layout.
      // lane row = l16; VGPR 0..3 hold K = 8*half + {0..7};
      // VGPR 4..7 hold K = 16 + 8*half + {0..7}  -> two 16B loads.
      v16bf aH[4], aL[4];
#pragma unroll
      for (int s = 0; s < 4; ++s) {
        const int m = mt * MTILE + wm * 64 + s * 16 + l16;
        const v8bf* pH =
            reinterpret_cast<const v8bf*>(gHi + (size_t)m * KPAD + kc * 32 + half * 8);
        const v8bf* pL =
            reinterpret_cast<const v8bf*>(gLo + (size_t)m * KPAD + kc * 32 + half * 8);
        const v8bf h0 = pH[0], h1 = pH[2];   // +16 bf16
        const v8bf l0 = pL[0], l1 = pL[2];
        aH[s] = __builtin_shufflevector(h0, h1, 0, 1, 2, 3, 4, 5, 6, 7,
                                        8, 9, 10, 11, 12, 13, 14, 15);
        aL[s] = __builtin_shufflevector(l0, l1, 0, 1, 2, 3, 4, 5, 6, 7,
                                        8, 9, 10, 11, 12, 13, 14, 15);
      }
      // B fragments (E^T cols): lane col = l16; VGPR 0..3 hold K = 8*half + {0..7};
      // VGPR 4..7 hold K = 16 + 8*half + {0..7}: two ds_load_b128.
      v16bf bH[2], bL[2];
#pragma unroll
      for (int u = 0; u < 2; ++u) {
        const int n = wn * 32 + u * 16 + l16;
        const v8bf* pH =
            reinterpret_cast<const v8bf*>(&ldsHi[n * LPITCH + kc * 32 + half * 8]);
        const v8bf* pL =
            reinterpret_cast<const v8bf*>(&ldsLo[n * LPITCH + kc * 32 + half * 8]);
        bH[u] = __builtin_shufflevector(pH[0], pH[2], 0, 1, 2, 3, 4, 5, 6, 7,
                                        8, 9, 10, 11, 12, 13, 14, 15);
        bL[u] = __builtin_shufflevector(pL[0], pL[2], 0, 1, 2, 3, 4, 5, 6, 7,
                                        8, 9, 10, 11, 12, 13, 14, 15);
      }
#pragma unroll
      for (int s = 0; s < 4; ++s)
#pragma unroll
        for (int u = 0; u < 2; ++u) {
          acc[s][u] = __builtin_amdgcn_wmma_f32_16x16x32_bf16(
              false, aH[s], false, bH[u], (short)0, acc[s][u], false, false);
          acc[s][u] = __builtin_amdgcn_wmma_f32_16x16x32_bf16(
              false, aH[s], false, bL[u], (short)0, acc[s][u], false, false);
          acc[s][u] = __builtin_amdgcn_wmma_f32_16x16x32_bf16(
              false, aL[s], false, bH[u], (short)0, acc[s][u], false, false);
        }
    }

    // ---- sigmoid + store (D layout: lane l -> N = l16, M = v + 8*half) ----
    // sigmoid via v_exp_f32 + v_rcp_f32 (~1 ulp, << split-bf16 GEMM noise);
    // non-temporal store: output is write-once, keep L2 for g reuse.
#pragma unroll
    for (int s = 0; s < 4; ++s)
#pragma unroll
      for (int u = 0; u < 2; ++u) {
        const int n = tileN + wn * 32 + u * 16 + l16;
        if (n < NE_) {
          const int mb = mt * MTILE + wm * 64 + s * 16 + 8 * half;
#pragma unroll
          for (int v = 0; v < 8; ++v) {
            const float x = acc[s][u][v];
            const float r = __builtin_amdgcn_rcpf(1.0f + __expf(-x));
            __builtin_nontemporal_store(r, &out[(size_t)(mb + v) * NE_ + n]);
          }
        }
      }
  }
}

// ---------------------------------------------------------------------------
extern "C" void kernel_launch(void* const* d_in, const int* in_sizes, int n_in,
                              void* d_out, int out_size, void* d_ws, size_t ws_size,
                              hipStream_t stream)
{
  (void)in_sizes; (void)n_in; (void)out_size; (void)ws_size;
  const int*   e1 = (const int*)d_in[0];
  const int*   ri = (const int*)d_in[1];
  const int*   ti = (const int*)d_in[2];
  const float* E  = (const float*)d_in[3];
  const float* R  = (const float*)d_in[4];
  const float* T  = (const float*)d_in[5];
  const float* F0 = (const float*)d_in[6];
  const float* F1 = (const float*)d_in[7];
  const float* F2 = (const float*)d_in[8];
  const float* F3 = (const float*)d_in[9];

  uint16_t* gHi = (uint16_t*)d_ws;                 // 1024*224 bf16
  uint16_t* gLo = gHi + (size_t)B_ * KPAD;         // 1024*224 bf16 (total ~0.9 MB)

  tuckercpd_prep<<<B_, KPAD, 0, stream>>>(e1, ri, ti, E, R, T,
                                          F0, F1, F2, F3, gHi, gLo);

  const int nblocks = (NE_ + NTILE - 1) / NTILE;   // 782
  tuckercpd_gemm<<<nblocks, 256, 0, stream>>>(E, gHi, gLo, (float*)d_out);
}
